// Net_73383811219612
// MI455X (gfx1250) — matrix operand order repaired and checked by reading, not compiled, not run on hardware
//
// MaxCutPool GNN forward on MI455X (gfx1250): all dense adjacency/feature GEMMs
// run through v_wmma_f32_16x16x32_f16 (f16 in, f32 accumulate). Adjacency,
// one-hot assignment and reachability masks are 0/1 -> exact in f16.
// Working set (A_h = 33.5MB) fits in the 192MB L2; problem is latency bound.
// Hot path: one wave32 computes a 16x32 output tile (2 WMMA accs sharing one
// A fragment), fragment loads are unconditional global_load_b128 (B is always
// consumed transposed so its fragment is contiguous in K). A tiny general
// kernel handles the ragged launches (Ka=3 / Nn=3 / Nn=1).
#include <hip/hip_runtime.h>

typedef _Float16 h16;
typedef __attribute__((ext_vector_type(16))) _Float16 v16h;
typedef __attribute__((ext_vector_type(8)))  float    v8f;

// ---------------------------------------------------------------------------
// Shared epilogue semantics:
//   out = act( alpha * (A @ B) / rowdiv + bias + Cadd )
//   act: 0 none, 1 relu, 2 tanh, 3 (v>0)?1:0
//   outputs (each optional): outF f32 [M,Nn], outH f16 [M,Nn], outHT f16 [Nn,M]
// ---------------------------------------------------------------------------

// ---------------- fast path: lda/ldbt %8==0, Ka%32==0, Nn%32==0 -------------
__global__ __launch_bounds__(32)
void wmma_gemm_fast(const h16* __restrict__ Ag, long sA, int lda,
                    const h16* __restrict__ BTg, long sBT, int ldbt,
                    const float* __restrict__ Cadd, long sC,
                    const float* __restrict__ bias,
                    const float* __restrict__ rowdiv, long sRd,
                    float alpha, int act,
                    float* __restrict__ outF, long sOF,
                    h16* __restrict__ outH, long sOH,
                    h16* __restrict__ outHT, long sOHT,
                    int M, int Nn, int Ka)
{
  const int bb   = blockIdx.z;
  const int tm   = blockIdx.x;
  const int tn0  = blockIdx.y * 2;          // two 16-wide column tiles per wave
  const int lane = threadIdx.x;
  const bool hi  = (lane >= 16);
  const int mA   = tm * 16 + (lane & 15);
  const int n0   = tn0 * 16 + (lane & 15);
  const h16* Arow = Ag + (long)bb * sA + (long)mA * lda;
  const h16* B0   = BTg + (long)bb * sBT + (long)n0 * ldbt;
  const h16* B1   = B0 + (long)16 * ldbt;
  const int kaOff = hi ? 16 : 0;   // A frag (16x32): one run of 16 K-values
  const int kbOff = hi ? 16 : 0;   // B frag (32x16): one run of 16 K-values

  v8f acc0 = {}, acc1 = {};
  union V { v16h v; uint4 q[2]; };
  for (int k0 = 0; k0 < Ka; k0 += 32) {
    __builtin_prefetch(Arow + k0 + 128, 0, 1);  // global_prefetch_b8 (speculative)
    V fa, f0, f1;
    fa.q[0] = *reinterpret_cast<const uint4*>(Arow + k0 + kaOff);
    fa.q[1] = *reinterpret_cast<const uint4*>(Arow + k0 + kaOff + 8);
    f0.q[0] = *reinterpret_cast<const uint4*>(B0 + k0 + kbOff);
    f0.q[1] = *reinterpret_cast<const uint4*>(B0 + k0 + kbOff + 8);
    f1.q[0] = *reinterpret_cast<const uint4*>(B1 + k0 + kbOff);
    f1.q[1] = *reinterpret_cast<const uint4*>(B1 + k0 + kbOff + 8);
    acc0 = __builtin_amdgcn_wmma_f32_16x16x32_f16(false, fa.v, false, f0.v,
                                                  (short)0, acc0, false, false);
    acc1 = __builtin_amdgcn_wmma_f32_16x16x32_f16(false, fa.v, false, f1.v,
                                                  (short)0, acc1, false, false);
  }

#pragma unroll
  for (int t = 0; t < 2; ++t) {
    const int n = n0 + t * 16;
    const v8f& acc = t ? acc1 : acc0;
#pragma unroll
    for (int r = 0; r < 8; ++r) {
      const int m = tm * 16 + r + (hi ? 8 : 0);  // C/D: VGPR r -> M=r (+8 hi half)
      float v = acc[r] * alpha;
      if (rowdiv) v /= fmaxf(rowdiv[(long)bb * sRd + m], 1e-12f);
      if (bias)   v += bias[n];
      if (Cadd)   v += Cadd[(long)bb * sC + (long)m * Nn + n];
      if (act == 1)      v = fmaxf(v, 0.f);
      else if (act == 2) v = tanhf(v);
      else if (act == 3) v = (v > 0.f) ? 1.f : 0.f;
      if (outF)  outF[(long)bb * sOF + (long)m * Nn + n] = v;
      if (outH)  outH[(long)bb * sOH + (long)m * Nn + n] = (h16)v;
      if (outHT) outHT[(long)bb * sOHT + (long)n * M + m] = (h16)v;
    }
  }
}

// ---------------- general path: guarded loads, used only for tiny launches --
__device__ __forceinline__ void load_run8g(v16h& f, int e0, const h16* __restrict__ row,
                                           int k, int Ka) {
#pragma unroll
  for (int t = 0; t < 8; ++t) f[e0 + t] = (k + t < Ka) ? row[k + t] : (h16)0.f;
}

__global__ __launch_bounds__(32)
void wmma_gemm_edge(const h16* __restrict__ Ag, long sA, int lda,
                    const h16* __restrict__ BTg, long sBT, int ldbt,
                    const float* __restrict__ Cadd, long sC,
                    const float* __restrict__ bias,
                    const float* __restrict__ rowdiv, long sRd,
                    float alpha, int act,
                    float* __restrict__ outF, long sOF,
                    h16* __restrict__ outH, long sOH,
                    h16* __restrict__ outHT, long sOHT,
                    int M, int Nn, int Ka)
{
  const int bb   = blockIdx.z;
  const int tm   = blockIdx.x;
  const int tn   = blockIdx.y;
  const int lane = threadIdx.x;
  const bool hi  = (lane >= 16);
  const int mA   = tm * 16 + (lane & 15);
  const int nB   = tn * 16 + (lane & 15);
  const bool colOK = (nB < Nn);
  const h16* Arow = Ag + (long)bb * sA + (long)mA * lda;
  const h16* Brow = BTg + (long)bb * sBT + (long)(colOK ? nB : 0) * ldbt;

  v8f acc = {};
  for (int k0 = 0; k0 < Ka; k0 += 32) {
    v16h af = {}, bf = {};
    const int ka = k0 + (hi ? 16 : 0);
    load_run8g(af, 0, Arow, ka, Ka);
    load_run8g(af, 8, Arow, ka + 8, Ka);
    if (colOK) {
      const int kb = k0 + (hi ? 16 : 0);
      load_run8g(bf, 0, Brow, kb, Ka);
      load_run8g(bf, 8, Brow, kb + 8, Ka);
    }
    acc = __builtin_amdgcn_wmma_f32_16x16x32_f16(false, af, false, bf,
                                                 (short)0, acc, false, false);
  }
  if (!colOK) return;
  const int n = nB;
#pragma unroll
  for (int r = 0; r < 8; ++r) {
    const int m = tm * 16 + r + (hi ? 8 : 0);
    float v = acc[r] * alpha;
    if (rowdiv) v /= fmaxf(rowdiv[(long)bb * sRd + m], 1e-12f);
    if (bias)   v += bias[n];
    if (Cadd)   v += Cadd[(long)bb * sC + (long)m * Nn + n];
    if (act == 1)      v = fmaxf(v, 0.f);
    else if (act == 2) v = tanhf(v);
    else if (act == 3) v = (v > 0.f) ? 1.f : 0.f;
    if (outF)  outF[(long)bb * sOF + (long)m * Nn + n] = v;
    if (outH)  outH[(long)bb * sOH + (long)m * Nn + n] = (h16)v;
    if (outHT) outHT[(long)bb * sOHT + (long)n * M + m] = (h16)v;
  }
}

// ---------------------------------------------------------------------------
// Glue kernels
// ---------------------------------------------------------------------------
__global__ void w_transpose_h(const float* __restrict__ W, int in, int out,
                              h16* __restrict__ WT, long total) {
  long i = (long)blockIdx.x * blockDim.x + threadIdx.x;
  if (i >= total) return;
  int o = (int)(i % out), r = (int)(i / out);
  WT[(long)o * in + r] = (h16)W[i];
}

__global__ void f32_to_h_rt(const float* __restrict__ src, h16* __restrict__ dstH,
                            h16* __restrict__ dstT, int M, int C, long total) {
  long i = (long)blockIdx.x * blockDim.x + threadIdx.x;
  if (i >= total) return;
  int c = (int)(i % C); long r = i / C; int n = (int)(r % M); int bb = (int)(r / M);
  float v = src[i];
  dstH[i] = (h16)v;
  dstT[(((long)bb * C) + c) * M + n] = (h16)v;
}

// f32 adjacency -> f16 copy (+optional binarized copy) + row degrees
__global__ void adj_prep(const float* __restrict__ A, int M, h16* __restrict__ Ah,
                         h16* __restrict__ AbH, float* __restrict__ deg) {
  const int r = blockIdx.x, bb = blockIdx.y;
  const float* row = A + ((long)bb * M + r) * M;
  h16* hrow = Ah + ((long)bb * M + r) * M;
  h16* brow = AbH ? AbH + ((long)bb * M + r) * M : nullptr;
  float s = 0.f;
  for (int j = threadIdx.x; j < M; j += blockDim.x) {
    float v = row[j]; s += v;
    hrow[j] = (h16)v;
    if (brow) brow[j] = (h16)(v > 0.f ? 1.f : 0.f);
  }
  __shared__ float sh[256];
  sh[threadIdx.x] = s; __syncthreads();
  for (int st = 128; st > 0; st >>= 1) {
    if (threadIdx.x < st) sh[threadIdx.x] += sh[threadIdx.x + st];
    __syncthreads();
  }
  if (threadIdx.x == 0) deg[(long)bb * M + r] = sh[0];
}

__global__ void batch_sum(const float* __restrict__ v, int M, float* __restrict__ out) {
  int bb = blockIdx.x;
  float s = 0.f;
  for (int i = threadIdx.x; i < M; i += blockDim.x) s += v[(long)bb * M + i];
  __shared__ float sh[256];
  sh[threadIdx.x] = s; __syncthreads();
  for (int st = 128; st > 0; st >>= 1) {
    if (threadIdx.x < st) sh[threadIdx.x] += sh[threadIdx.x + st];
    __syncthreads();
  }
  if (threadIdx.x == 0) out[bb] = sh[0];
}

// BatchNorm (training mode) stats over all B*M rows, per channel
__global__ void bn_stats(const float* __restrict__ x, int rows, int C,
                         float* __restrict__ mean, float* __restrict__ inv) {
  const int c = blockIdx.x;
  float s = 0.f, q = 0.f;
  for (int r = threadIdx.x; r < rows; r += blockDim.x) {
    float v = x[(long)r * C + c]; s += v; q += v * v;
  }
  __shared__ float ssum[256], ssq[256];
  ssum[threadIdx.x] = s; ssq[threadIdx.x] = q; __syncthreads();
  for (int st = 128; st > 0; st >>= 1) {
    if (threadIdx.x < st) { ssum[threadIdx.x] += ssum[threadIdx.x + st];
                            ssq[threadIdx.x]  += ssq[threadIdx.x + st]; }
    __syncthreads();
  }
  if (threadIdx.x == 0) {
    float m = ssum[0] / rows;
    float v = ssq[0] / rows - m * m;
    mean[c] = m; inv[c] = rsqrtf(v + 1e-5f);
  }
}

// BN-apply, writing into the 192-wide concat buffers (f32 + f16 + f16^T)
__global__ void bn_apply(const float* __restrict__ x, const float* __restrict__ mean,
                         const float* __restrict__ inv, const float* __restrict__ g,
                         const float* __restrict__ be,
                         int M, int C, int catC, int off,
                         float* __restrict__ catF, h16* __restrict__ catH,
                         h16* __restrict__ catT, long total) {
  long i = (long)blockIdx.x * blockDim.x + threadIdx.x;
  if (i >= total) return;
  int c = (int)(i % C); long r = i / C; int n = (int)(r % M); int bb = (int)(r / M);
  float v = (x[i] - mean[c]) * inv[c] * g[c] + be[c];
  long o = (((long)bb * M) + n) * catC + off + c;
  catF[o] = v; catH[o] = (h16)v;
  catT[(((long)bb * catC) + off + c) * M + n] = (h16)v;
}

// cut[b] = s^T A s / max(sum(A), 1e-12)
__global__ void cut_loss(const float* __restrict__ Afull, const float* __restrict__ s,
                         const float* __restrict__ sumA, int M, float* __restrict__ cut) {
  const int bb = blockIdx.x;
  __shared__ float sc[512];
  __shared__ float red[256];
  for (int i = threadIdx.x; i < M; i += blockDim.x) sc[i] = s[(long)bb * M + i];
  __syncthreads();
  float acc = 0.f;
  const long tot = (long)M * M;
  const float* Ab = Afull + (long)bb * M * M;
  for (long e = threadIdx.x; e < tot; e += blockDim.x) {
    int i = (int)(e / M), j = (int)(e % M);
    acc += sc[i] * Ab[e] * sc[j];
  }
  red[threadIdx.x] = acc; __syncthreads();
  for (int st = 128; st > 0; st >>= 1) {
    if (threadIdx.x < st) red[threadIdx.x] += red[threadIdx.x + st];
    __syncthreads();
  }
  if (threadIdx.x == 0) cut[bb] = red[0] / fmaxf(sumA[bb], 1e-12f);
}

// top_k by rank counting (stable: ties broken by index, matches lax.top_k)
__global__ void topk_rank(const float* __restrict__ s, int M, int K,
                          float* __restrict__ vals, int* __restrict__ idx) {
  const int bb = blockIdx.x;
  __shared__ float sh[512];
  for (int i = threadIdx.x; i < M; i += blockDim.x) sh[i] = s[(long)bb * M + i];
  __syncthreads();
  for (int i = threadIdx.x; i < M; i += blockDim.x) {
    float si = sh[i]; int rank = 0;
    for (int j = 0; j < M; ++j) {
      float sj = sh[j];
      rank += (sj > si) || ((sj == si) && (j < i));
    }
    if (rank < K) { vals[(long)bb * K + rank] = si; idx[(long)bb * K + rank] = i; }
  }
}

__global__ void reach_init(const int* __restrict__ idx, int M, int K,
                           float* __restrict__ Rf, h16* __restrict__ Rt) {
  const int k = blockIdx.x, bb = blockIdx.y;
  const int src = idx[(long)bb * K + k];
  for (int n = threadIdx.x; n < M; n += blockDim.x) {
    float v = (n == src) ? 1.f : 0.f;
    Rf[(((long)bb * M) + n) * K + k] = v;
    Rt[(((long)bb * K) + k) * M + n] = (h16)v;
  }
}

// assign = first reached supernode (vals sorted desc => argmax of masked vals)
__global__ void assign_k(const float* __restrict__ Rf, int K,
                         int* __restrict__ asg, long total) {
  long i = (long)blockIdx.x * blockDim.x + threadIdx.x;
  if (i >= total) return;
  const float* row = Rf + i * (long)K;
  int a = 0;
  for (int k = K - 1; k >= 0; --k) if (row[k] > 0.f) a = k;
  asg[i] = a;
}

// S^T one-hot: ST[b,k,n] = (assign[b,n]==k)
__global__ void build_st(const int* __restrict__ asg, int M, int K,
                         h16* __restrict__ ST, long total) {
  long i = (long)blockIdx.x * blockDim.x + threadIdx.x;
  if (i >= total) return;
  int n = (int)(i % M); long r = i / M; int k = (int)(r % K); int bb = (int)(r / K);
  ST[i] = (h16)((asg[(long)bb * M + n] == k) ? 1.f : 0.f);
}

// x_pool = x[idx] * vals  (write f16 row-major + f16 transposed)
__global__ void gather_pool(const float* __restrict__ cat, const int* __restrict__ idx,
                            const float* __restrict__ vals, int K, int M, int C,
                            h16* __restrict__ xpH, h16* __restrict__ xpT, long total) {
  long i = (long)blockIdx.x * blockDim.x + threadIdx.x;
  if (i >= total) return;
  int c = (int)(i % C); long r = i / C; int k = (int)(r % K); int bb = (int)(r / K);
  int src = idx[(long)bb * K + k];
  float v = cat[(((long)bb * M) + src) * C + c] * vals[(long)bb * K + k];
  xpH[i] = (h16)v;
  xpT[(((long)bb * C) + c) * K + k] = (h16)v;
}

// mean readout -> lin1(relu) -> lin2 -> log_softmax (tiny, one block per graph)
__global__ void readout(const float* __restrict__ cat, int M, int C,
                        const float* __restrict__ W1, const float* __restrict__ b1,
                        const float* __restrict__ W2, const float* __restrict__ b2,
                        float* __restrict__ out) {
  const int bb = blockIdx.x;
  __shared__ float g[192];
  __shared__ float hbuf[64];
  for (int c = threadIdx.x; c < C; c += blockDim.x) {
    float s = 0.f;
    for (int n = 0; n < M; ++n) s += cat[(((long)bb * M) + n) * C + c];
    g[c] = s / M;
  }
  __syncthreads();
  for (int j = threadIdx.x; j < 64; j += blockDim.x) {
    float s = b1[j];
    for (int c = 0; c < C; ++c) s += g[c] * W1[c * 64 + j];
    hbuf[j] = fmaxf(s, 0.f);
  }
  __syncthreads();
  if (threadIdx.x == 0) {
    float l0 = b2[0], l1 = b2[1];
    for (int j = 0; j < 64; ++j) { l0 += hbuf[j] * W2[j * 2]; l1 += hbuf[j] * W2[j * 2 + 1]; }
    float mx = fmaxf(l0, l1);
    float lse = mx + logf(expf(l0 - mx) + expf(l1 - mx));
    out[bb * 2 + 0] = l0 - lse;
    out[bb * 2 + 1] = l1 - lse;
  }
}

__global__ void final_loss(const float* __restrict__ c1, const float* __restrict__ c2,
                           int Bsz, float* __restrict__ out) {
  __shared__ float sh[64];
  float s = (threadIdx.x < Bsz) ? (c1[threadIdx.x] + c2[threadIdx.x]) : 0.f;
  sh[threadIdx.x] = s; __syncthreads();
  for (int st = 32; st > 0; st >>= 1) {
    if (threadIdx.x < st) sh[threadIdx.x] += sh[threadIdx.x + st];
    __syncthreads();
  }
  if (threadIdx.x == 0) out[2 * Bsz] = sh[0] / Bsz;  // d_out[128] = l1 + l2
}

// ---------------------------------------------------------------------------
// Host-side orchestration
// ---------------------------------------------------------------------------
static inline void gemm(hipStream_t st, int Bsz, int M, int Nn, int Ka,
                        const h16* A, long sA, int lda,
                        const h16* BT, long sBT, int ldbt,
                        const float* Cadd, long sC,
                        const float* bias,
                        const float* rowdiv, long sRd,
                        float alpha, int act,
                        float* outF, long sOF,
                        h16* outH, long sOH,
                        h16* outHT, long sOHT) {
  const bool fast = ((Ka & 31) == 0) && ((lda & 7) == 0) && ((ldbt & 7) == 0) &&
                    ((Nn & 31) == 0);
  if (fast) {
    dim3 grid(M / 16, Nn / 32, Bsz);
    wmma_gemm_fast<<<grid, 32, 0, st>>>(A, sA, lda, BT, sBT, ldbt, Cadd, sC, bias,
                                        rowdiv, sRd, alpha, act, outF, sOF, outH, sOH,
                                        outHT, sOHT, M, Nn, Ka);
  } else {
    dim3 grid(M / 16, (Nn + 15) / 16, Bsz);
    wmma_gemm_edge<<<grid, 32, 0, st>>>(A, sA, lda, BT, sBT, ldbt, Cadd, sC, bias,
                                        rowdiv, sRd, alpha, act, outF, sOF, outH, sOH,
                                        outHT, sOHT, M, Nn, Ka);
  }
}

struct GnnW {
  const h16* WlT[3]; const h16* WrT[3];
  const float* bias[3]; const float* gam[3]; const float* bet[3];
};
struct PoolW {
  const h16* WmpT; const h16* Wm1T; const h16* Wm2T;
  const float* bmp; const float* bm1; const float* bm2;
};

// One gnn_block: 3x (SAGE conv -> relu -> BN) with 192-wide concat output.
static void run_gnn(hipStream_t st, int Bn, int M,
                    const h16* Ah, const float* deg,
                    const h16* xH0, int ldx0, long sxH0,
                    const h16* xT0, long sxT0, int Cin0,
                    const GnnW& w,
                    float* catF, h16* catH, h16* catT,
                    h16* aggH, float* tmpU, float* convF,
                    float* bnM, float* bnI) {
  const h16* curH = xH0; int curLd = ldx0; long scurH = sxH0;
  const h16* curT = xT0; long scurT = sxT0; int curC = Cin0;
  for (int c = 0; c < 3; ++c) {
    const int Cout = 64;
    // mean_agg = (A @ x) / deg  -> f16 [M, curC]
    gemm(st, Bn, M, curC, M, Ah, (long)M * M, M, curT, scurT, M,
         nullptr, 0, nullptr, deg, M, 1.f, 0,
         nullptr, 0, aggH, (long)M * curC, nullptr, 0);
    // u = x @ Wr + b -> f32
    gemm(st, Bn, M, Cout, curC, curH, scurH, curLd, w.WrT[c], 0, curC,
         nullptr, 0, w.bias[c], nullptr, 0, 1.f, 0,
         tmpU, (long)M * Cout, nullptr, 0, nullptr, 0);
    // conv = relu(agg @ Wl + u) -> f32
    gemm(st, Bn, M, Cout, curC, aggH, (long)M * curC, curC, w.WlT[c], 0, curC,
         tmpU, (long)M * Cout, nullptr, nullptr, 0, 1.f, 1,
         convF, (long)M * Cout, nullptr, 0, nullptr, 0);
    bn_stats<<<Cout, 256, 0, st>>>(convF, Bn * M, Cout, bnM, bnI);
    long tot = (long)Bn * M * Cout;
    bn_apply<<<(int)((tot + 255) / 256), 256, 0, st>>>(
        convF, bnM, bnI, w.gam[c], w.bet[c], M, Cout, 192, c * 64,
        catF, catH, catT, tot);
    curH = catH + c * 64; curLd = 192; scurH = (long)M * 192;
    curT = catT + (long)c * 64 * M; scurT = (long)192 * M; curC = 64;
  }
}

// MaxCutPool: ScoreNet, cut loss, top-k, 3-hop reach, assignment, S^T A S, gather.
static void run_pool(hipStream_t st, int Bn, int M, int K,
                     const h16* Ah, const h16* AbH, const float* deg,
                     const float* Afull, const float* sumA,
                     const float* catF, const h16* catT,
                     const PoolW& pw,
                     h16* heterH, h16* h1H, h16* h2H, float* sF,
                     float* vals, int* idx,
                     float* Rf0, float* Rf1, h16* Rt0, h16* Rt1,
                     int* asg, h16* ST, h16* AST,
                     float* Apool, float* cut,
                     h16* xpH, h16* xpT) {
  const long sAh = (long)M * M;
  // hetero MP: x - DELTA * mean_agg(A, x)   (DELTA = 2.0)
  gemm(st, Bn, M, 192, M, Ah, sAh, M, catT, (long)192 * M, M,
       catF, (long)M * 192, nullptr, deg, M, -2.0f, 0,
       nullptr, 0, heterH, (long)M * 192, nullptr, 0);
  gemm(st, Bn, M, 64, 192, heterH, (long)M * 192, 192, pw.WmpT, 0, 192,
       nullptr, 0, pw.bmp, nullptr, 0, 1.f, 1,
       nullptr, 0, h1H, (long)M * 64, nullptr, 0);
  gemm(st, Bn, M, 32, 64, h1H, (long)M * 64, 64, pw.Wm1T, 0, 64,
       nullptr, 0, pw.bm1, nullptr, 0, 1.f, 1,
       nullptr, 0, h2H, (long)M * 32, nullptr, 0);
  gemm(st, Bn, M, 1, 32, h2H, (long)M * 32, 32, pw.Wm2T, 0, 32,
       nullptr, 0, pw.bm2, nullptr, 0, 1.f, 2,
       sF, (long)M, nullptr, 0, nullptr, 0);
  cut_loss<<<Bn, 256, 0, st>>>(Afull, sF, sumA, M, cut);
  topk_rank<<<Bn, 256, 0, st>>>(sF, M, K, vals, idx);
  reach_init<<<dim3(K, Bn), 256, 0, st>>>(idx, M, K, Rf0, Rt0);
  // reach = ((reach + Ab @ reach) > 0), 3 hops, ping-pong to avoid RAW races
  float* Rfi = Rf0; float* Rfo = Rf1; h16* Rti = Rt0; h16* Rto = Rt1;
  for (int it = 0; it < 3; ++it) {
    gemm(st, Bn, M, K, M, AbH, sAh, M, Rti, (long)K * M, M,
         Rfi, (long)M * K, nullptr, nullptr, 0, 1.f, 3,
         Rfo, (long)M * K, nullptr, 0, Rto, (long)K * M);
    float* tf = Rfi; Rfi = Rfo; Rfo = tf;
    h16* th = Rti; Rti = Rto; Rto = th;
  }
  long totNM = (long)Bn * M;
  assign_k<<<(int)((totNM + 255) / 256), 256, 0, st>>>(Rfi, K, asg, totNM);
  long totST = (long)Bn * K * M;
  build_st<<<(int)((totST + 255) / 256), 256, 0, st>>>(asg, M, K, ST, totST);
  // AS^T = (A @ S)^T, then A_pool = S^T @ AS
  gemm(st, Bn, M, K, M, Ah, sAh, M, ST, (long)K * M, M,
       nullptr, 0, nullptr, nullptr, 0, 1.f, 0,
       nullptr, 0, nullptr, 0, AST, (long)K * M);
  gemm(st, Bn, K, K, M, ST, (long)K * M, M, AST, (long)K * M, M,
       nullptr, 0, nullptr, nullptr, 0, 1.f, 0,
       Apool, (long)K * K, nullptr, 0, nullptr, 0);
  long totXP = (long)Bn * K * 192;
  gather_pool<<<(int)((totXP + 255) / 256), 256, 0, st>>>(
      catF, idx, vals, K, M, 192, xpH, xpT, totXP);
}

extern "C" void kernel_launch(void* const* d_in, const int* in_sizes, int n_in,
                              void* d_out, int out_size, void* d_ws, size_t ws_size,
                              hipStream_t stream) {
  (void)in_sizes; (void)n_in; (void)out_size; (void)ws_size;
  hipStream_t st = stream;
  constexpr int Bn = 64, N1 = 512, K1 = 256, K2 = 128, EMB = 192;
  const float* x   = (const float*)d_in[0];
  const float* Ain = (const float*)d_in[1];
  auto P = [&](int i) { return (const float*)d_in[i]; };

  // bump allocator over workspace (deterministic layout each call)
  char* base = (char*)d_ws; size_t off = 0;
  auto alloc  = [&](size_t bytes) -> void* {
    off = (off + 255) & ~(size_t)255; void* p = base + off; off += bytes; return p;
  };
  auto allocH = [&](long n) { return (h16*)alloc((size_t)n * sizeof(h16)); };
  auto allocF = [&](long n) { return (float*)alloc((size_t)n * sizeof(float)); };
  auto allocI = [&](long n) { return (int*)alloc((size_t)n * sizeof(int)); };
  auto wt = [&](const float* W, int in, int out) -> h16* {
    h16* d = allocH((long)in * out);
    long tot = (long)in * out;
    w_transpose_h<<<(int)((tot + 255) / 256), 256, 0, st>>>(W, in, out, d, tot);
    return d;
  };

  // ---- parameters (setup_inputs dict order): x, A, gnn1..3, pool1..2, lin1..2
  GnnW gw[3];
  const int gb[3] = {2, 17, 32};
  for (int g = 0; g < 3; ++g) {
    const int cins[3] = {(g == 0) ? 3 : EMB, 64, 64};
    for (int c = 0; c < 3; ++c) {
      int ib = gb[g] + c * 5;
      gw[g].WlT[c]  = wt(P(ib + 0), cins[c], 64);
      gw[g].WrT[c]  = wt(P(ib + 1), cins[c], 64);
      gw[g].bias[c] = P(ib + 2);
      gw[g].gam[c]  = P(ib + 3);
      gw[g].bet[c]  = P(ib + 4);
    }
  }
  PoolW pw[2];
  const int pb[2] = {47, 53};
  for (int p = 0; p < 2; ++p) {
    pw[p].WmpT = wt(P(pb[p] + 0), EMB, 64); pw[p].bmp = P(pb[p] + 1);
    pw[p].Wm1T = wt(P(pb[p] + 2), 64, 32);  pw[p].bm1 = P(pb[p] + 3);
    pw[p].Wm2T = wt(P(pb[p] + 4), 32, 1);   pw[p].bm2 = P(pb[p] + 5);
  }
  const float* W1 = P(59); const float* b1 = P(60);
  const float* W2 = P(61); const float* b2 = P(62);

  // ---- level-1 inputs
  h16* xH = allocH((long)Bn * N1 * 3);
  h16* xT = allocH((long)Bn * 3 * N1);
  { long tot = (long)Bn * N1 * 3;
    f32_to_h_rt<<<(int)((tot + 255) / 256), 256, 0, st>>>(x, xH, xT, N1, 3, tot); }
  h16* A1h = allocH((long)Bn * N1 * N1);
  float* deg1 = allocF((long)Bn * N1);
  adj_prep<<<dim3(N1, Bn), 256, 0, st>>>(Ain, N1, A1h, nullptr, deg1);  // A is 0/1 => Ab==A
  float* sumA1 = allocF(Bn);
  batch_sum<<<Bn, 256, 0, st>>>(deg1, N1, sumA1);

  // ---- shared scratch (sized for level 1, reused by levels 2/3)
  h16*   aggH  = allocH((long)Bn * N1 * EMB);
  float* tmpU  = allocF((long)Bn * N1 * 64);
  float* convF = allocF((long)Bn * N1 * 64);
  float* bnM = allocF(64); float* bnI = allocF(64);
  h16* h1H = allocH((long)Bn * N1 * 64);
  h16* h2H = allocH((long)Bn * N1 * 32);
  float* sF = allocF((long)Bn * N1);
  float* vals = allocF((long)Bn * K1); int* idxB = allocI((long)Bn * K1);
  float* Rf0 = allocF((long)Bn * N1 * K1); float* Rf1 = allocF((long)Bn * N1 * K1);
  h16* Rt0 = allocH((long)Bn * K1 * N1);   h16* Rt1 = allocH((long)Bn * K1 * N1);
  int* asg = allocI((long)Bn * N1);
  h16* ST  = allocH((long)Bn * K1 * N1);
  h16* AST = allocH((long)Bn * K1 * N1);

  // ---- GNN 1 (n=512)
  float* cat1F = allocF((long)Bn * N1 * EMB);
  h16* cat1H = allocH((long)Bn * N1 * EMB);
  h16* cat1T = allocH((long)Bn * EMB * N1);
  run_gnn(st, Bn, N1, A1h, deg1, xH, 3, (long)N1 * 3, xT, (long)3 * N1, 3,
          gw[0], cat1F, cat1H, cat1T, aggH, tmpU, convF, bnM, bnI);

  // ---- Pool 1 (512 -> 256)
  float* Apool1 = allocF((long)Bn * K1 * K1);
  float* cut1 = allocF(Bn);
  h16* xp1H = allocH((long)Bn * K1 * EMB);
  h16* xp1T = allocH((long)Bn * EMB * K1);
  run_pool(st, Bn, N1, K1, A1h, A1h, deg1, Ain, sumA1, cat1F, cat1T, pw[0],
           aggH, h1H, h2H, sF, vals, idxB, Rf0, Rf1, Rt0, Rt1, asg, ST, AST,
           Apool1, cut1, xp1H, xp1T);

  // ---- level-2 adjacency
  h16* A2h  = allocH((long)Bn * K1 * K1);
  h16* Ab2h = allocH((long)Bn * K1 * K1);
  float* deg2 = allocF((long)Bn * K1);
  adj_prep<<<dim3(K1, Bn), 256, 0, st>>>(Apool1, K1, A2h, Ab2h, deg2);
  float* sumA2 = allocF(Bn);
  batch_sum<<<Bn, 256, 0, st>>>(deg2, K1, sumA2);

  // ---- GNN 2 (n=256)
  float* cat2F = allocF((long)Bn * K1 * EMB);
  h16* cat2H = allocH((long)Bn * K1 * EMB);
  h16* cat2T = allocH((long)Bn * EMB * K1);
  run_gnn(st, Bn, K1, A2h, deg2, xp1H, EMB, (long)K1 * EMB, xp1T, (long)EMB * K1,
          EMB, gw[1], cat2F, cat2H, cat2T, aggH, tmpU, convF, bnM, bnI);

  // ---- Pool 2 (256 -> 128)
  float* Apool2 = allocF((long)Bn * K2 * K2);
  float* cut2 = allocF(Bn);
  h16* xp2H = allocH((long)Bn * K2 * EMB);
  h16* xp2T = allocH((long)Bn * EMB * K2);
  run_pool(st, Bn, K1, K2, A2h, Ab2h, deg2, Apool1, sumA2, cat2F, cat2T, pw[1],
           aggH, h1H, h2H, sF, vals, idxB, Rf0, Rf1, Rt0, Rt1, asg, ST, AST,
           Apool2, cut2, xp2H, xp2T);

  // ---- level-3 adjacency
  h16* A3h = allocH((long)Bn * K2 * K2);
  float* deg3 = allocF((long)Bn * K2);
  adj_prep<<<dim3(K2, Bn), 256, 0, st>>>(Apool2, K2, A3h, nullptr, deg3);

  // ---- GNN 3 (n=128)
  float* cat3F = allocF((long)Bn * K2 * EMB);
  h16* cat3H = allocH((long)Bn * K2 * EMB);
  h16* cat3T = allocH((long)Bn * EMB * K2);
  run_gnn(st, Bn, K2, A3h, deg3, xp2H, EMB, (long)K2 * EMB, xp2T, (long)EMB * K2,
          EMB, gw[2], cat3F, cat3H, cat3T, aggH, tmpU, convF, bnM, bnI);

  // ---- readout + losses: d_out[0..127] log-probs, d_out[128] = l1 + l2
  float* out = (float*)d_out;
  readout<<<Bn, 192, 0, st>>>(cat3F, K2, EMB, W1, b1, W2, b2, out);
  final_loss<<<1, 64, 0, st>>>(cut1, cut2, Bn, out);
}
